// PhysicsAttentionBase_49435073577101
// MI455X (gfx1250) — compile-verified, matrix operand order checked by reading, not run on hardware
//
#include <hip/hip_runtime.h>

// ---------------------------------------------------------------------------
// Transolver physics attention for gfx1250 (MI455X), f16 WMMA + f32 accum.
// B=4, N=32768, DIM=256, HEADS=8, DIM_HEAD=32, SLICE_NUM=64
// ---------------------------------------------------------------------------

typedef _Float16 v16h __attribute__((ext_vector_type(16)));
typedef _Float16 v8h  __attribute__((ext_vector_type(8)));
typedef _Float16 v4h  __attribute__((ext_vector_type(4)));
typedef float    v8f  __attribute__((ext_vector_type(8)));

union U8f { v8f v; float f[8]; };
union U8h { v8h v; _Float16 f[8]; };

#define TOKENS   131072L   // B*N

// ---- workspace layout (bytes) ----
#define WS_WXH    0UL                      // 256x256 f16 packed  (131072)
#define WS_WFXH   131072UL
#define WS_WOUTH  262144UL
#define WS_WSH    393216UL                 // 32x64 f16 packed    (4096)
#define WS_NUMER  397312UL                 // B*H*S*D f32         (262144)
#define WS_DENOM  659456UL                 // B*H*S   f32         (8192)
#define WS_TOKT   667648UL                 // B*H*D*S f16         (131072)
#define WS_XMID   798720UL                 // TOKENS*256 f16      (67108864)
#define WS_FX     67907584UL               // TOKENS*256 f16      (67108864)

// ---------------------------------------------------------------------------
// WMMA helpers
// ---------------------------------------------------------------------------
__device__ __forceinline__ v8f wmma_f16(v16h a, v16h b, v8f c) {
  return __builtin_amdgcn_wmma_f32_16x16x32_f16(false, a, false, b,
                                                (short)0, c, false, false);
}

// A-fragment (16x32 f16) from a row-major f16 LDS tile.
// strideH (in halves) must be a multiple of 8 so rows stay 16B aligned.
// Lane l<16 holds row M=l with K = ktB/2 + {0..7, 16..23};
// lane l>=16 holds row M=l-16 with K = ktB/2 + {8..15, 24..31}.
__device__ __forceinline__ v16h frag_a(const _Float16* tileBase, int strideH,
                                       int lane, int ktB /*bytes = kt*64*/) {
  int r  = lane & 15;
  int hi = (lane & 16) ? 16 : 0;
  const char* p = (const char*)(tileBase + r * strideH);
  v8h lo = *(const v8h*)(p + ktB + hi);
  v8h hv = *(const v8h*)(p + ktB + 32 + hi);
  return __builtin_shufflevector(lo, hv, 0, 1, 2, 3, 4, 5, 6, 7,
                                 8, 9, 10, 11, 12, 13, 14, 15);
}

// B-fragment from packed weights: fragment `frag` is 32 lanes x 16 halves
// contiguous (32 bytes/lane) -> two coalesced b128 global loads.
__device__ __forceinline__ v16h frag_w(const _Float16* Wp, int frag, int lane) {
  const v8h* p = (const v8h*)Wp + ((size_t)frag * 32 + lane) * 2;
  v8h lo = p[0];
  v8h hv = p[1];
  return __builtin_shufflevector(lo, hv, 0, 1, 2, 3, 4, 5, 6, 7,
                                 8, 9, 10, 11, 12, 13, 14, 15);
}

// One wave computes a 16(M) x 128(N) strip of a 64x256 GEMM, K=256.
// All 8 B-fragments of a K-step are fetched into distinct registers before
// the 8 WMMAs so the L2 latency is paid once per K-step, not per WMMA.
__device__ __forceinline__ void gemm_tile8(const _Float16* sA, int strideH,
                                           const _Float16* Wp, int mt, int nh,
                                           int lane, v8f acc[8]) {
#pragma unroll
  for (int j = 0; j < 8; ++j)
    acc[j] = (v8f){0.f, 0.f, 0.f, 0.f, 0.f, 0.f, 0.f, 0.f};
  const _Float16* tb = sA + mt * 16 * strideH;
#pragma unroll
  for (int kt = 0; kt < 8; ++kt) {
    v16h a = frag_a(tb, strideH, lane, kt * 64);
    v16h b[8];
#pragma unroll
    for (int j = 0; j < 8; ++j)
      b[j] = frag_w(Wp, kt * 16 + nh * 8 + j, lane);
#pragma unroll
    for (int j = 0; j < 8; ++j)
      acc[j] = wmma_f16(a, b[j], acc[j]);
  }
}

// ---------------------------------------------------------------------------
// K0: pack weights to fragment-ready f16, zero accumulators
// ---------------------------------------------------------------------------
__global__ __launch_bounds__(256) void k_prep(
    const float* __restrict__ Wx, const float* __restrict__ Wfx,
    const float* __restrict__ Wout, const float* __restrict__ Wslice,
    _Float16* __restrict__ Wxh, _Float16* __restrict__ Wfxh,
    _Float16* __restrict__ Wouth, _Float16* __restrict__ Wsh,
    float* __restrict__ numer, float* __restrict__ denom) {
  int t = blockIdx.x * 256 + threadIdx.x;
  if (t < 12288) {                       // pack the three 256x256 matrices
    int mat  = t / 4096;
    int r    = t % 4096;                 // (kt*16+nt)*32 + lane
    int lane = r & 31;
    int tile = r >> 5;
    int kt   = tile >> 4;
    int nt   = tile & 15;
    const float* W = (mat == 0) ? Wx : ((mat == 1) ? Wfx : Wout);
    _Float16*    D = (mat == 0) ? Wxh : ((mat == 1) ? Wfxh : Wouth);
    int n  = nt * 16 + (lane & 15);
    int kb = (lane & 16) ? 8 : 0;
#pragma unroll
    for (int i = 0; i < 16; ++i) {
      int k = kt * 32 + ((i < 8) ? i : (i + 8)) + kb;
      D[(size_t)r * 16 + i] = (_Float16)W[n * 256 + k];   // B(k,n)=W[n][k]
    }
  } else if (t < 12416) {                // pack Wslice (K=32, N=64 -> NT=4)
    int r    = t - 12288;
    int lane = r & 31;
    int nt   = r >> 5;
    int n    = nt * 16 + (lane & 15);
    int kb   = (lane & 16) ? 8 : 0;
#pragma unroll
    for (int i = 0; i < 16; ++i) {
      int k = ((i < 8) ? i : (i + 8)) + kb;
      Wsh[(size_t)r * 16 + i] = (_Float16)Wslice[n * 32 + k];
    }
  } else if (t < 12416 + 65536) {
    numer[t - 12416] = 0.f;
  } else if (t < 12416 + 65536 + 2048) {
    denom[t - 12416 - 65536] = 0.f;
  }
}

// ---------------------------------------------------------------------------
// K1: x -> x_mid (f16), fx (f16).  64-token tile / block, 8 waves.
// ---------------------------------------------------------------------------
__device__ __forceinline__ void store_half_tiles(v8f acc[8],
                                                 _Float16* __restrict__ dst,
                                                 const float* __restrict__ bias,
                                                 long t0, int mt, int nh,
                                                 int lane) {
  int hi8 = (lane & 16) ? 8 : 0;
  int nlo = lane & 15;
#pragma unroll
  for (int j = 0; j < 8; ++j) {
    int   n  = (nh * 8 + j) * 16 + nlo;
    float bv = bias[n];
    U8f u; u.v = acc[j];
#pragma unroll
    for (int r = 0; r < 8; ++r) {
      long m = t0 + mt * 16 + hi8 + r;
      dst[m * 256 + n] = (_Float16)(u.f[r] + bv);
    }
  }
}

__global__ __launch_bounds__(256) void k_midfx(
    const float* __restrict__ x, const _Float16* __restrict__ Wxh,
    const _Float16* __restrict__ Wfxh, const float* __restrict__ bx,
    const float* __restrict__ bfx, _Float16* __restrict__ xmid,
    _Float16* __restrict__ fx) {
  __shared__ _Float16 sX[64 * 264];      // 33792 B, rows 528B (16B aligned)
  int  tid = threadIdx.x;
  long t0  = (long)blockIdx.x * 64;
#pragma unroll
  for (int it = 0; it < 16; ++it) {      // stage 64x256 f32 -> f16 LDS
    int f  = tid + it * 256;
    int m  = f >> 6;
    int c4 = f & 63;
    float4 v = *(const float4*)(x + (t0 + m) * 256 + c4 * 4);
    v4h h;
    h[0] = (_Float16)v.x; h[1] = (_Float16)v.y;
    h[2] = (_Float16)v.z; h[3] = (_Float16)v.w;
    *(v4h*)&sX[m * 264 + c4 * 4] = h;
  }
  __syncthreads();
  int lane = tid & 31, wave = tid >> 5;
  int mt = wave & 3, nh = wave >> 2;
  v8f acc[8];
  gemm_tile8(sX, 264, Wxh, mt, nh, lane, acc);
  store_half_tiles(acc, xmid, bx, t0, mt, nh, lane);
  gemm_tile8(sX, 264, Wfxh, mt, nh, lane, acc);
  store_half_tiles(acc, fx, bfx, t0, mt, nh, lane);
}

// ---------------------------------------------------------------------------
// K2: slice pooling, fully on the WMMA pipe.
// Per head: proj (WMMA) -> softmax (writes w TRANSPOSED [s][m]) ->
// pooling numer = w^T @ fx as WMMA with fx staged transposed [d][m].
// Each wave accumulates one 16x16 numer tile across 8 sub-tiles (512 tokens),
// then flushes with f32 global atomics.
// ---------------------------------------------------------------------------
__global__ __launch_bounds__(256) void k_pool(
    const _Float16* __restrict__ xmid, const _Float16* __restrict__ fx,
    const _Float16* __restrict__ Wsh, const float* __restrict__ bslice,
    const float* __restrict__ temp, float* __restrict__ numer,
    float* __restrict__ denom) {
  __shared__ _Float16 sXm[64 * 40];      // x_mid head slice  [m][d]
  __shared__ _Float16 sFxT[32 * 72];     // fx head slice, transposed [d][m]
  __shared__ float    sProj[64 * 66];
  __shared__ _Float16 sWT[64 * 72];      // slice weights, transposed [s][m]
  int  tid = threadIdx.x, lane = tid & 31, wave = tid >> 5;
  long t0 = (long)blockIdx.x * 512;
  int  b  = (int)(t0 >> 15);

  for (int h = 0; h < 8; ++h) {
    float invt = 1.0f / fminf(fmaxf(temp[h], 0.5f), 5.0f);
    // this wave's numer tile: rows s = st*16.., cols d = dt*16..
    int st = wave >> 1, dt = wave & 1;
    v8f  pacc = (v8f){0.f, 0.f, 0.f, 0.f, 0.f, 0.f, 0.f, 0.f};
    float dsum = 0.f;                    // valid for tid<64 (s = tid)
    for (int sub = 0; sub < 8; ++sub) {
      long tt = t0 + sub * 64;
      {
        int m = tid >> 2, q = tid & 3;   // 64 rows x 4 x 16B chunks
        *(v8h*)&sXm[m * 40 + q * 8] =
            *(const v8h*)&xmid[(tt + m) * 256 + h * 32 + q * 8];
        U8h f8;
        f8.v = *(const v8h*)&fx[(tt + m) * 256 + h * 32 + q * 8];
#pragma unroll
        for (int i = 0; i < 8; ++i)       // transpose into [d][m]
          sFxT[(q * 8 + i) * 72 + m] = f8.f[i];
      }
      __syncthreads();
      // proj GEMM: 16 tiles / 8 waves
#pragma unroll
      for (int ti = 0; ti < 2; ++ti) {
        int tile = wave * 2 + ti;
        int mt = tile >> 2, nt = tile & 3;
        v16h a = frag_a(sXm + mt * 16 * 40, 40, lane, 0);
        v16h bb = frag_w(Wsh, nt, lane);
        v8f  c = (v8f){0.f, 0.f, 0.f, 0.f, 0.f, 0.f, 0.f, 0.f};
        c = wmma_f16(a, bb, c);
        U8f u; u.v = c;
        int hi8 = (lane & 16) ? 8 : 0;
#pragma unroll
        for (int r = 0; r < 8; ++r)
          sProj[(mt * 16 + hi8 + r) * 66 + nt * 16 + (lane & 15)] = u.f[r];
      }
      __syncthreads();
      if (tid < 64) {                    // softmax over slices, token m=tid
        int   m  = tid;
        float mx = -1e30f;
        for (int s = 0; s < 64; ++s) {
          float v = (sProj[m * 66 + s] + bslice[s]) * invt;
          sProj[m * 66 + s] = v;
          mx = fmaxf(mx, v);
        }
        float sum = 0.f;
        for (int s = 0; s < 64; ++s) {
          float e = __expf(sProj[m * 66 + s] - mx);
          sProj[m * 66 + s] = e;
          sum += e;
        }
        float rs = 1.f / sum;
        for (int s = 0; s < 64; ++s)     // transposed store: w^T[s][m]
          sWT[s * 72 + m] = (_Float16)(sProj[m * 66 + s] * rs);
      }
      __syncthreads();
      // pooling: numer += w^T @ fx ; one 16x16 tile / wave, K=64 (2 steps)
#pragma unroll
      for (int kt = 0; kt < 2; ++kt) {
        v16h a  = frag_a(sWT + st * 16 * 72, 72, lane, kt * 64);
        v16h bb = frag_a(sFxT + dt * 16 * 72, 72, lane, kt * 64);
        pacc = wmma_f16(a, bb, pacc);
      }
      if (tid < 64) {                    // denom: row-sum of w^T (s = tid)
        for (int m8 = 0; m8 < 8; ++m8) {
          U8h wrow; wrow.v = *(const v8h*)&sWT[tid * 72 + m8 * 8];
#pragma unroll
          for (int i = 0; i < 8; ++i) dsum += (float)wrow.f[i];
        }
      }
      __syncthreads();
    }
    // flush this head's partials
    {
      U8f u; u.v = pacc;
      int hi8 = (lane & 16) ? 8 : 0;
      int d   = dt * 16 + (lane & 15);
#pragma unroll
      for (int r = 0; r < 8; ++r) {
        int s = st * 16 + hi8 + r;
        atomicAdd(&numer[(((size_t)b * 8 + h) * 64 + s) * 32 + d], u.f[r]);
      }
      if (tid < 64)
        atomicAdd(&denom[((size_t)b * 8 + h) * 64 + tid], dsum);
    }
  }
}

// ---------------------------------------------------------------------------
// K3: tiny 64-slice attention per (b,h).  Writes out_tok transposed (d-major)
// so K4 can load it directly as WMMA B-fragments.
// ---------------------------------------------------------------------------
__global__ __launch_bounds__(64) void k_attn(
    const float* __restrict__ numer, const float* __restrict__ denom,
    const float* __restrict__ Wqkv, _Float16* __restrict__ tokT) {
  __shared__ float sc[64][65];
  __shared__ float kb[64][65];
  __shared__ float vb[64][65];
  int s  = threadIdx.x;
  int bh = blockIdx.x;                   // b*8 + h
  float st[32], q[32];
  float inv = 1.f / (denom[(size_t)bh * 64 + s] + 0.01f);
#pragma unroll
  for (int d = 0; d < 32; ++d)
    st[d] = numer[((size_t)bh * 64 + s) * 32 + d] * inv;
#pragma unroll
  for (int e = 0; e < 32; ++e) {
    float aq = 0.f, ak = 0.f, av = 0.f;
#pragma unroll
    for (int d = 0; d < 32; ++d) {
      aq += st[d] * Wqkv[e * 32 + d];
      ak += st[d] * Wqkv[(32 + e) * 32 + d];
      av += st[d] * Wqkv[(64 + e) * 32 + d];
    }
    q[e] = aq; kb[s][e] = ak; vb[s][e] = av;
  }
  __syncthreads();
  float mx = -1e30f;
  for (int j = 0; j < 64; ++j) {
    float d0 = 0.f;
#pragma unroll
    for (int d = 0; d < 32; ++d) d0 += q[d] * kb[j][d];
    d0 *= 0.17677669529663687f;          // 32^-0.5
    sc[s][j] = d0;
    mx = fmaxf(mx, d0);
  }
  float sum = 0.f, o[32];
#pragma unroll
  for (int d = 0; d < 32; ++d) o[d] = 0.f;
  for (int j = 0; j < 64; ++j) {
    float e = __expf(sc[s][j] - mx);
    sum += e;
#pragma unroll
    for (int d = 0; d < 32; ++d) o[d] += e * vb[j][d];
  }
  float rs = 1.f / sum;
#pragma unroll
  for (int d = 0; d < 32; ++d)
    tokT[((size_t)bh * 32 + d) * 64 + s] = (_Float16)(o[d] * rs);
}

// ---------------------------------------------------------------------------
// K4: recompute slice weights, unslice (WMMA), final 256x256 GEMM (WMMA)
// ---------------------------------------------------------------------------
__global__ __launch_bounds__(256) void k_out(
    const _Float16* __restrict__ xmid, const _Float16* __restrict__ tokT,
    const _Float16* __restrict__ Wsh, const _Float16* __restrict__ Wouth,
    const float* __restrict__ bslice, const float* __restrict__ temp,
    const float* __restrict__ bout, float* __restrict__ out) {
  __shared__ _Float16 sTmp[64 * 40];     // xmid slice OR out_tok^T (32x72)
  __shared__ float    sProj[64 * 66];
  __shared__ _Float16 sW[64 * 72];       // slice weights [m][s]
  __shared__ _Float16 sOx[64 * 264];     // out_x accumulator, f16
  int  tid = threadIdx.x, lane = tid & 31, wave = tid >> 5;
  long t0 = (long)blockIdx.x * 64;
  int  b  = (int)(t0 >> 15);

  for (int h = 0; h < 8; ++h) {
    {
      int m = tid >> 2, q = tid & 3;     // stage x_mid head slice
      *(v8h*)&sTmp[m * 40 + q * 8] =
          *(const v8h*)&xmid[(t0 + m) * 256 + h * 32 + q * 8];
    }
    __syncthreads();
    float invt = 1.0f / fminf(fmaxf(temp[h], 0.5f), 5.0f);
    // proj GEMM (reads sTmp); then sTmp is free to be reused for out_tok^T
    {
#pragma unroll
      for (int ti = 0; ti < 2; ++ti) {
        int tile = wave * 2 + ti;
        int mt = tile >> 2, nt = tile & 3;
        v16h a = frag_a(sTmp + mt * 16 * 40, 40, lane, 0);
        v16h bb = frag_w(Wsh, nt, lane);
        v8f  c = (v8f){0.f, 0.f, 0.f, 0.f, 0.f, 0.f, 0.f, 0.f};
        c = wmma_f16(a, bb, c);
        U8f u; u.v = c;
        int hi8 = (lane & 16) ? 8 : 0;
#pragma unroll
        for (int r = 0; r < 8; ++r)
          sProj[(mt * 16 + hi8 + r) * 66 + nt * 16 + (lane & 15)] = u.f[r];
      }
    }
    __syncthreads();
    {                                    // stage out_tok^T (32x64 halves)
      int d = tid >> 3, q = tid & 7;
      *(v8h*)&sTmp[d * 72 + q * 8] =
          *(const v8h*)&tokT[((size_t)(b * 8 + h) * 32 + d) * 64 + q * 8];
    }
    if (tid < 64) {                      // softmax (same math as k_pool)
      int   m  = tid;
      float mx = -1e30f;
      for (int s = 0; s < 64; ++s) {
        float v = (sProj[m * 66 + s] + bslice[s]) * invt;
        sProj[m * 66 + s] = v;
        mx = fmaxf(mx, v);
      }
      float sum = 0.f;
      for (int s = 0; s < 64; ++s) {
        float e = __expf(sProj[m * 66 + s] - mx);
        sProj[m * 66 + s] = e;
        sum += e;
      }
      float rs = 1.f / sum;
      for (int s = 0; s < 64; ++s)
        sW[m * 72 + s] = (_Float16)(sProj[m * 66 + s] * rs);
    }
    __syncthreads();
    // unslice GEMM: (64xS) @ (SxD) per head, one 16x16 tile per wave, K=64
    {
      int mt = wave >> 1, nt = wave & 1;
      v8f c = (v8f){0.f, 0.f, 0.f, 0.f, 0.f, 0.f, 0.f, 0.f};
#pragma unroll
      for (int kt = 0; kt < 2; ++kt) {
        v16h a  = frag_a(sW + mt * 16 * 72, 72, lane, kt * 64);
        v16h bb = frag_a(sTmp + nt * 16 * 72, 72, lane, kt * 64); // d-major
        c = wmma_f16(a, bb, c);
      }
      U8f u; u.v = c;
      int hi8 = (lane & 16) ? 8 : 0;
#pragma unroll
      for (int r = 0; r < 8; ++r)
        sOx[(mt * 16 + hi8 + r) * 264 + h * 32 + nt * 16 + (lane & 15)] =
            (_Float16)u.f[r];
    }
    __syncthreads();
  }
  // final GEMM: out = out_x @ Wout^T + bout
  int mt = wave & 3, nh = wave >> 2;
  v8f acc[8];
  gemm_tile8(sOx, 264, Wouth, mt, nh, lane, acc);
  int hi8 = (lane & 16) ? 8 : 0;
#pragma unroll
  for (int j = 0; j < 8; ++j) {
    int   n  = (nh * 8 + j) * 16 + (lane & 15);
    float bv = bout[n];
    U8f u; u.v = acc[j];
#pragma unroll
    for (int r = 0; r < 8; ++r) {
      long m = t0 + mt * 16 + hi8 + r;
      out[m * 256 + n] = u.f[r] + bv;
    }
  }
}

// ---------------------------------------------------------------------------
extern "C" void kernel_launch(void* const* d_in, const int* in_sizes, int n_in,
                              void* d_out, int out_size, void* d_ws,
                              size_t ws_size, hipStream_t stream) {
  (void)in_sizes; (void)n_in; (void)out_size; (void)ws_size;
  const float* x      = (const float*)d_in[0];
  const float* Wx     = (const float*)d_in[1];
  const float* bx     = (const float*)d_in[2];
  const float* Wfx    = (const float*)d_in[3];
  const float* bfx    = (const float*)d_in[4];
  const float* Wslice = (const float*)d_in[5];
  const float* bslice = (const float*)d_in[6];
  const float* Wqkv   = (const float*)d_in[7];
  const float* Wout   = (const float*)d_in[8];
  const float* bout   = (const float*)d_in[9];
  const float* temp   = (const float*)d_in[10];
  float* out = (float*)d_out;

  char* ws = (char*)d_ws;
  _Float16* Wxh   = (_Float16*)(ws + WS_WXH);
  _Float16* Wfxh  = (_Float16*)(ws + WS_WFXH);
  _Float16* Wouth = (_Float16*)(ws + WS_WOUTH);
  _Float16* Wsh   = (_Float16*)(ws + WS_WSH);
  float*    numer = (float*)(ws + WS_NUMER);
  float*    denom = (float*)(ws + WS_DENOM);
  _Float16* tokT  = (_Float16*)(ws + WS_TOKT);
  _Float16* xmid  = (_Float16*)(ws + WS_XMID);
  _Float16* fx    = (_Float16*)(ws + WS_FX);

  k_prep<<<313, 256, 0, stream>>>(Wx, Wfx, Wout, Wslice, Wxh, Wfxh, Wouth,
                                  Wsh, numer, denom);
  k_midfx<<<2048, 256, 0, stream>>>(x, Wxh, Wfxh, bx, bfx, xmid, fx);
  k_pool<<<256, 256, 0, stream>>>(xmid, fx, Wsh, bslice, temp, numer, denom);
  k_attn<<<32, 64, 0, stream>>>(numer, denom, Wqkv, tokT);
  k_out<<<2048, 256, 0, stream>>>(xmid, tokT, Wsh, Wouth, bslice, temp, bout,
                                  out);
}
